// MoeGate_15710990369658
// MI455X (gfx1250) — matrix operand order repaired and checked by reading, not compiled, not run on hardware
//
#include <hip/hip_runtime.h>
#include <hip/hip_bf16.h>

#define T_TOKENS 16384
#define B_BATCH  4
#define S_SEQ    4096
#define H_DIM    2048
#define E_EXP    64
#define K_TOP    8
#define ALPHA_C  0.01f
#define EPS_NORM 1e-20f

typedef __attribute__((ext_vector_type(16))) __bf16 v16bf;
typedef __attribute__((ext_vector_type(8)))  __bf16 v8bf;
typedef __attribute__((ext_vector_type(8)))  float  v8f;

// Workspace layout (floats): [0,256) counts[B][E]; [256,512) pisum[B][E];
// then bf16 weight copy (E*H elements) starting at byte offset 2048.

__global__ void moe_ws_init(float* ws) {
    int i = blockIdx.x * blockDim.x + threadIdx.x;
    if (i < 2 * B_BATCH * E_EXP) ws[i] = 0.0f;
}

__global__ void moe_wcvt(const float* __restrict__ w, __bf16* __restrict__ wbf) {
    int i = blockIdx.x * blockDim.x + threadIdx.x;   // E*H = 131072 threads
    wbf[i] = (__bf16)w[i];
}

__global__ __launch_bounds__(256) void moe_gate_main(
    const float*  __restrict__ x,
    const __bf16* __restrict__ wbf,
    float* __restrict__ out_idx,
    float* __restrict__ out_w,
    float* __restrict__ ws) {

    __shared__ float lds_logits[8][16][E_EXP];   // 32 KB: per-wave 16 tokens x 64 experts
    __shared__ float lds_cnt[E_EXP];
    __shared__ float lds_pi[E_EXP];

    const int tid  = threadIdx.x;
    const int wave = tid >> 5;
    const int lane = tid & 31;
    const int half = lane >> 4;      // 0: lanes 0-15, 1: lanes 16-31
    const int lid  = lane & 15;
    const int tokBase = blockIdx.x * 128 + wave * 16;

    if (tid < E_EXP) { lds_cnt[tid] = 0.0f; lds_pi[tid] = 0.0f; }

    // ---------- Phase 1: logits via v_wmma_f32_16x16x32_bf16, double-buffered ----------
    v8f acc[4];
    #pragma unroll
    for (int et = 0; et < 4; ++et)
        #pragma unroll
        for (int j = 0; j < 8; ++j) acc[et][j] = 0.0f;

    const float*  xrow = x   + (size_t)(tokBase + lid) * H_DIM;
    const __bf16* wcol = wbf + (size_t)lid * H_DIM + half * 16;  // expert row lid (+16*et later)

    // Chunk loader: A raw f32 (16 values) + B bf16 (4 expert tiles x 16 values).
    auto load_chunk = [&](int kb, float4 a[4], v8bf b[4][2]) {
        const int k0 = kb + half * 16;          // A: lanes<16 K=kb..kb+15, lanes>=16 +16
        const float4* pa  = reinterpret_cast<const float4*>(xrow + k0);
        a[0] = pa[0]; a[1] = pa[1];
        const float4* pa2 = reinterpret_cast<const float4*>(xrow + k0 + 8);
        a[2] = pa2[0]; a[3] = pa2[1];
        #pragma unroll
        for (int et = 0; et < 4; ++et) {        // B: lanes<16 K=kb..kb+15, lanes>=16 +16
            const __bf16* wr = wcol + (size_t)et * 16 * H_DIM + kb;
            b[et][0] = *reinterpret_cast<const v8bf*>(wr);
            b[et][1] = *reinterpret_cast<const v8bf*>(wr + 8);
        }
    };

    auto compute_chunk = [&](const float4 a[4], const v8bf b[4][2]) {
        v16bf av;
        av[0]=(__bf16)a[0].x;  av[1]=(__bf16)a[0].y;  av[2]=(__bf16)a[0].z;  av[3]=(__bf16)a[0].w;
        av[4]=(__bf16)a[1].x;  av[5]=(__bf16)a[1].y;  av[6]=(__bf16)a[1].z;  av[7]=(__bf16)a[1].w;
        av[8]=(__bf16)a[2].x;  av[9]=(__bf16)a[2].y;  av[10]=(__bf16)a[2].z; av[11]=(__bf16)a[2].w;
        av[12]=(__bf16)a[3].x; av[13]=(__bf16)a[3].y; av[14]=(__bf16)a[3].z; av[15]=(__bf16)a[3].w;
        #pragma unroll
        for (int et = 0; et < 4; ++et) {
            v16bf bv = __builtin_shufflevector(b[et][0], b[et][1],
                         0,1,2,3,4,5,6,7,8,9,10,11,12,13,14,15);
            acc[et] = __builtin_amdgcn_wmma_f32_16x16x32_bf16(
                          false, av, false, bv, (short)0, acc[et], false, false);
        }
    };

    float4 aP[4], aQ[4];
    v8bf   bP[4][2], bQ[4][2];

    load_chunk(0, aP, bP);                       // prologue
    for (int kb = 0; kb < H_DIM; kb += 64) {     // 32 iterations, 2 chunks each
        __builtin_prefetch(xrow + kb + 1024, 0, 1);   // global_prefetch: x stream ~4KB ahead
        load_chunk(kb + 32, aQ, bQ);             // next chunk in flight while P computes
        compute_chunk(aP, bP);
        if (kb + 64 < H_DIM) load_chunk(kb + 64, aP, bP);
        compute_chunk(aQ, bQ);
    }

    // C layout: lanes 0-15 -> M=j, lanes 16-31 -> M=j+8; N = lid. Col = et*16 + lid.
    #pragma unroll
    for (int j = 0; j < 8; ++j) {
        const int m = j + half * 8;
        #pragma unroll
        for (int et = 0; et < 4; ++et)
            lds_logits[wave][m][et * 16 + lid] = acc[et][j];
    }
    __syncthreads();

    // ---------- Phase 2: softmax + top-8 per token (lanes 0-15 of each wave) ----------
    if (half == 0) {
        const int token = tokBase + lid;
        float* r = lds_logits[wave][lid];

        float mx = -3.4e38f;
        for (int e = 0; e < E_EXP; ++e) mx = fmaxf(mx, r[e]);
        float sum = 0.0f;
        for (int e = 0; e < E_EXP; ++e) { float v = __expf(r[e] - mx); r[e] = v; sum += v; }
        const float inv = 1.0f / sum;
        for (int e = 0; e < E_EXP; ++e) {
            float p = r[e] * inv;
            r[e] = p;
            atomicAdd(&lds_pi[e], p);          // pi accumulation (mean over S later)
        }

        float tw[K_TOP]; int ti[K_TOP]; float wsum = 0.0f;
        #pragma unroll
        for (int i = 0; i < K_TOP; ++i) {
            float bm = -1.0f; int bi = 0;
            for (int e = 0; e < E_EXP; ++e) {
                float v = r[e];
                if (v > bm) { bm = v; bi = e; } // strict > keeps lowest index on ties
            }
            r[bi] = -2.0f;
            tw[i] = bm; ti[i] = bi; wsum += bm;
            atomicAdd(&lds_cnt[bi], 1.0f);     // expert counts
        }
        const float dn = 1.0f / (wsum + EPS_NORM);
        #pragma unroll
        for (int i = 0; i < K_TOP; ++i) {
            out_idx[(size_t)token * K_TOP + i] = (float)ti[i];
            out_w  [(size_t)token * K_TOP + i] = tw[i] * dn;
        }
    }
    __syncthreads();

    // ---------- Phase 3: block-level stats -> global workspace ----------
    if (tid < E_EXP) {
        const int b = tokBase / S_SEQ;         // 128-token block never crosses batch
        atomicAdd(&ws[b * E_EXP + tid], lds_cnt[tid]);
        atomicAdd(&ws[B_BATCH * E_EXP + b * E_EXP + tid], lds_pi[tid]);
    }
}

__global__ void moe_loss_final(const float* __restrict__ ws, float* __restrict__ out_loss) {
    if (threadIdx.x == 0 && blockIdx.x == 0) {
        float accb = 0.0f;
        for (int b = 0; b < B_BATCH; ++b) {
            float s = 0.0f;
            for (int e = 0; e < E_EXP; ++e) {
                float fi = ws[b * E_EXP + e] * ((float)E_EXP / (float)(S_SEQ * K_TOP));
                float pi = ws[B_BATCH * E_EXP + b * E_EXP + e] / (float)S_SEQ;
                s += pi * fi;
            }
            accb += s;
        }
        out_loss[0] = (accb / (float)B_BATCH) * ALPHA_C;
    }
}

extern "C" void kernel_launch(void* const* d_in, const int* in_sizes, int n_in,
                              void* d_out, int out_size, void* d_ws, size_t ws_size,
                              hipStream_t stream) {
    const float* x = (const float*)d_in[0];     // [4, 4096, 2048] f32
    const float* w = (const float*)d_in[1];     // [64, 2048] f32

    float* out      = (float*)d_out;
    float* out_idx  = out;                                    // [T,8] (as float)
    float* out_w    = out + (size_t)T_TOKENS * K_TOP;         // [T,8]
    float* out_loss = out + 2 * (size_t)T_TOKENS * K_TOP;     // scalar

    float*  ws  = (float*)d_ws;
    __bf16* wbf = (__bf16*)((char*)d_ws + 2 * B_BATCH * E_EXP * sizeof(float));

    moe_ws_init <<<1, 512, 0, stream>>>(ws);
    moe_wcvt    <<<(E_EXP * H_DIM) / 256, 256, 0, stream>>>(w, wbf);
    moe_gate_main<<<T_TOKENS / 128, 256, 0, stream>>>(x, wbf, out_idx, out_w, ws);
    moe_loss_final<<<1, 1, 0, stream>>>(ws, out_loss);
}
